// MemoryNet_32023276159359
// MI455X (gfx1250) — matrix operand order repaired and checked by reading, not compiled, not run on hardware
//
#include <hip/hip_runtime.h>
#include <hip/hip_bf16.h>

typedef __attribute__((ext_vector_type(16))) __bf16 v16bf;
typedef __attribute__((ext_vector_type(8)))  __bf16 v8bf;
typedef __attribute__((ext_vector_type(8)))  float  v8f;

#define N_MEM   256
#define N_EMB   128
#define ROWS_TOTAL (16 * 4096)   // B*S = 65536 query rows

union FragBF { v16bf v; v8bf h[2]; };

// ---------------------------------------------------------------------------
// Kernel 1: key = memory@Wk^T+bk, value = memory@Wv^T+bv.
// Store keyT[e][m] (bf16) = B-fragment feed for logits GEMM (K=e, N=m),
// and value[m][e] (bf16) = B-fragment feed for hidden GEMM (K=m, N=e).
// 128 blocks x 256 threads: one (m,e) output pair per thread.
// ---------------------------------------------------------------------------
__global__ void prep_kv_kernel(const float* __restrict__ mem,
                               const float* __restrict__ Wk,
                               const float* __restrict__ bk,
                               const float* __restrict__ Wv,
                               const float* __restrict__ bv,
                               __bf16* __restrict__ keyT,   // [128][256]
                               __bf16* __restrict__ val)    // [256][128]
{
    int idx = blockIdx.x * blockDim.x + threadIdx.x;   // 0..32767
    int m = idx >> 7;         // 0..255
    int e = idx & 127;        // 0..127
    const float* mr  = mem + m * N_EMB;
    const float* wkr = Wk  + e * N_EMB;
    const float* wvr = Wv  + e * N_EMB;
    float ak = 0.f, av = 0.f;
#pragma unroll 8
    for (int i = 0; i < N_EMB; ++i) {
        float x = mr[i];
        ak = __builtin_fmaf(x, wkr[i], ak);
        av = __builtin_fmaf(x, wvr[i], av);
    }
    keyT[e * N_MEM + m] = (__bf16)(ak + bk[e]);
    val [m * N_EMB + e] = (__bf16)(av + bv[e]);
}

// ---------------------------------------------------------------------------
// Kernel 2: per wave: 16 query rows.
//   logits[16,256] = Q(bf16) @ keyT      (16 m-tiles x 4 k-steps WMMA)
//   softmax over m (f32, in accumulator registers, 16-lane xor reductions)
//   hidden[16,128] = P(bf16) @ value     (8 e-tiles x 8 k-steps WMMA)
// LDS: keyT 64KB (reused for P staging after barrier) + value 64KB = 128KB.
// ---------------------------------------------------------------------------
__global__ void __launch_bounds__(256)
memnet_attn_kernel(const float* __restrict__ input,     // [65536][128] f32
                   const __bf16* __restrict__ keyT,     // [128][256] bf16
                   const __bf16* __restrict__ val,      // [256][128] bf16
                   float* __restrict__ out)             // [65536][128] f32
{
    __shared__ __align__(16) __bf16 sKeyT[N_EMB * N_MEM];   // 64KB, later: P staging
    __shared__ __align__(16) __bf16 sVal [N_MEM * N_EMB];   // 64KB

    const int tid  = threadIdx.x;
    const int lane = tid & 31;
    const int wave = tid >> 5;            // 0..7
    const int mrow = lane & 15;           // M index inside fragment
    const int kh   = lane >> 4;           // K-half select (A layout)

    // ---- cooperative LDS fill (keyT + value), uint4 copies ----
    {
        const uint4* gK = (const uint4*)keyT;
        const uint4* gV = (const uint4*)val;
        uint4* sK = (uint4*)sKeyT;
        uint4* sV = (uint4*)sVal;
#pragma unroll
        for (int i = 0; i < 16; ++i) {               // 4096 uint4 per array
            sK[tid + 256 * i] = gK[tid + 256 * i];
            sV[tid + 256 * i] = gV[tid + 256 * i];
        }
    }
    __syncthreads();

    const long qrow0 = (long)blockIdx.x * 128 + wave * 16;  // this wave's 16 rows

    // ---- load Q tile as 4 bf16 A-fragments (16x32 each) ----
    // A layout: lane m holds k = 32c + 8*kh + e (e<8) and 32c + 16 + 8*kh + (e-8)
    const float* qrow = input + (qrow0 + mrow) * N_EMB;
    v16bf aQ[4];
#pragma unroll
    for (int c = 0; c < 4; ++c) {
        const int b0 = 32 * c + 8 * kh;
        const int b1 = 32 * c + 16 + 8 * kh;
        float4 f0 = *(const float4*)(qrow + b0);
        float4 f1 = *(const float4*)(qrow + b0 + 4);
        float4 f2 = *(const float4*)(qrow + b1);
        float4 f3 = *(const float4*)(qrow + b1 + 4);
        FragBF u;
        u.v[0]=(__bf16)f0.x; u.v[1]=(__bf16)f0.y; u.v[2]=(__bf16)f0.z; u.v[3]=(__bf16)f0.w;
        u.v[4]=(__bf16)f1.x; u.v[5]=(__bf16)f1.y; u.v[6]=(__bf16)f1.z; u.v[7]=(__bf16)f1.w;
        u.v[8]=(__bf16)f2.x; u.v[9]=(__bf16)f2.y; u.v[10]=(__bf16)f2.z; u.v[11]=(__bf16)f2.w;
        u.v[12]=(__bf16)f3.x; u.v[13]=(__bf16)f3.y; u.v[14]=(__bf16)f3.z; u.v[15]=(__bf16)f3.w;
        aQ[c] = u.v;
    }

    // ---- GEMM1: logits[16,256] in 16 accumulator tiles ----
    // B layout (32x16): lane L holds column n = 16t + (L&15), k in table order (as A).
    v8f acc[16];
#pragma unroll
    for (int t = 0; t < 16; ++t) {
        v8f a = {};
#pragma unroll
        for (int c = 0; c < 4; ++c) {
            FragBF b;
#pragma unroll
            for (int s = 0; s < 16; ++s)
                b.v[s] = sKeyT[(32 * c + 8 * kh + (s & 7) + 16 * (s >> 3)) * N_MEM + 16 * t + mrow];
            a = __builtin_amdgcn_wmma_f32_16x16x32_bf16(
                    false, aQ[c], false, b.v, (short)0, a, false, false);
        }
        acc[t] = a;
    }

    // ---- softmax over m (per query row = per accumulator slot r) ----
    // C layout: lane holds n=(lane&15)+16t, VGPR r holds row m = r + 8*kh.
#pragma unroll
    for (int r = 0; r < 8; ++r) {
        float mx = acc[0][r];
#pragma unroll
        for (int t = 1; t < 16; ++t) mx = fmaxf(mx, acc[t][r]);
#pragma unroll
        for (int off = 1; off < 16; off <<= 1)
            mx = fmaxf(mx, __shfl_xor(mx, off, 32));   // 16-lane groups only
        float s = 0.f;
#pragma unroll
        for (int t = 0; t < 16; ++t) {
            float ev = __expf(acc[t][r] - mx);
            acc[t][r] = ev;
            s += ev;
        }
#pragma unroll
        for (int off = 1; off < 16; off <<= 1)
            s += __shfl_xor(s, off, 32);
        float inv = 1.0f / s;
#pragma unroll
        for (int t = 0; t < 16; ++t) acc[t][r] *= inv;
    }

    // ---- stage P (bf16) to LDS, reusing keyT region (barrier first) ----
    __syncthreads();                       // everyone done reading sKeyT
    __bf16* pw = sKeyT + wave * (16 * N_MEM);   // this wave's 16x256 tile
#pragma unroll
    for (int t = 0; t < 16; ++t) {
        const int col = (lane & 15) + 16 * t;
#pragma unroll
        for (int r = 0; r < 8; ++r)
            pw[(r + 8 * kh) * N_MEM + col] = (__bf16)acc[t][r];
    }
    // same-wave DS ordering: stores complete before our own loads below.

    // ---- GEMM2: hidden[16,128] = P @ value ----
    v8f hacc[8];
#pragma unroll
    for (int t = 0; t < 8; ++t) hacc[t] = (v8f){};
#pragma unroll
    for (int c = 0; c < 8; ++c) {          // K chunks over m (32 each)
        const __bf16* pr = pw + mrow * N_MEM;
        FragBF a;
        a.h[0] = *(const v8bf*)(pr + 32 * c + 8 * kh);
        a.h[1] = *(const v8bf*)(pr + 32 * c + 16 + 8 * kh);
#pragma unroll
        for (int t = 0; t < 8; ++t) {      // N tiles over e (16 each)
            FragBF b;
#pragma unroll
            for (int s = 0; s < 16; ++s)
                b.v[s] = sVal[(32 * c + 8 * kh + (s & 7) + 16 * (s >> 3)) * N_EMB + 16 * t + mrow];
            hacc[t] = __builtin_amdgcn_wmma_f32_16x16x32_bf16(
                          false, a.v, false, b.v, (short)0, hacc[t], false, false);
        }
    }

    // ---- write hidden (f32): 16 lanes cover 16 consecutive e -> coalesced ----
    float* orow = out + qrow0 * N_EMB;
#pragma unroll
    for (int t = 0; t < 8; ++t) {
        const int e = (lane & 15) + 16 * t;
#pragma unroll
        for (int r = 0; r < 8; ++r)
            orow[(r + 8 * kh) * N_EMB + e] = hacc[t][r];
    }
}

extern "C" void kernel_launch(void* const* d_in, const int* in_sizes, int n_in,
                              void* d_out, int out_size, void* d_ws, size_t ws_size,
                              hipStream_t stream) {
    const float* input  = (const float*)d_in[0];   // [16,4096,128]
    const float* memory = (const float*)d_in[1];   // [256,128]
    const float* Wk     = (const float*)d_in[2];   // [128,128]
    const float* bk     = (const float*)d_in[3];   // [128]
    const float* Wv     = (const float*)d_in[4];   // [128,128]
    const float* bv     = (const float*)d_in[5];   // [128]
    float* out = (float*)d_out;

    __bf16* keyT = (__bf16*)d_ws;                                  // 64KB
    __bf16* val  = (__bf16*)((char*)d_ws + N_EMB * N_MEM * 2);     // 64KB

    prep_kv_kernel<<<128, 256, 0, stream>>>(memory, Wk, bk, Wv, bv, keyT, val);
    memnet_attn_kernel<<<ROWS_TOTAL / 128, 256, 0, stream>>>(input, keyT, val, out);
}
